// SelfAttention_60206851555982
// MI455X (gfx1250) — matrix operand order repaired and checked by reading, not compiled, not run on hardware
//
#include <hip/hip_runtime.h>
#include <math.h>

// ---------------------------------------------------------------------------
// Types for CDNA5 WMMA (wave32, v_wmma_f32_16x16x32_bf16)
// ---------------------------------------------------------------------------
typedef __attribute__((ext_vector_type(16))) __bf16 v16bf;
typedef __attribute__((ext_vector_type(8)))  float  v8f;
typedef __attribute__((ext_vector_type(4)))  int    v4i;

union FragB16 { v16bf v; uint4 u[2]; };

// CDNA5 async global->LDS copy path (ASYNCcnt-tracked), guarded so the code
// still compiles (register-copy fallback) if the toolchain lacks the builtins.
#if defined(__AMDGCN__) && defined(__gfx1250__) && \
    __has_builtin(__builtin_amdgcn_global_load_async_to_lds_b128) && \
    __has_builtin(__builtin_amdgcn_s_wait_asynccnt)
#define USE_ASYNC_LDS 1
#else
#define USE_ASYNC_LDS 0
#endif

#if USE_ASYNC_LDS
typedef __attribute__((address_space(1))) v4i* as1_v4i;
typedef __attribute__((address_space(3))) v4i* as3_v4i;
__device__ __forceinline__ void async_copy_b128(const void* g, void* lds) {
  __builtin_amdgcn_global_load_async_to_lds_b128((as1_v4i)g, (as3_v4i)lds, 0, 0);
}
#endif

__device__ __forceinline__ unsigned short f32_to_bf16(float f) {
  unsigned int u = __float_as_uint(f);
  u += 0x7FFFu + ((u >> 16) & 1u);   // round-to-nearest-even
  return (unsigned short)(u >> 16);
}

#define BMt 128
#define BNt 128
#define BKt 32
#define LDKp 40   // padded LDS row stride (bf16 elems): 80B, 16B-aligned rows

// ---------------------------------------------------------------------------
// Generic batched NT GEMM:  C[m][n] = sum_k A[m][k] * B[n][k]
//  - AF32/BF32: operand is fp32 in memory, converted to bf16 at LDS-fill time
//    (bscale folded into B conversion); bf16 operands use async LDS copies.
//  - MODE 0: store fp32 row-major (ldc, + bz*cBS)
//  - MODE 1: store bf16, remap (m,n)->Q/K layout [b*H+h][t][d]
//  - MODE 2: store bf16, remap (m,n)->V^T layout [b*H+h][d][t]
//  - MODE 3: store bf16, remap per-batch (t,d)->attn-concat [b][t][h*DH+d]
// 8 waves (256 thr); each wave computes a 32x64 patch = 2x4 tiles of 16x16.
// All dims are multiples of the tile sizes for this problem -> no bounds checks.
// ---------------------------------------------------------------------------
template<bool AF32, bool BF32, int MODE>
__global__ __launch_bounds__(256) void gemm_wmma(
    const void* __restrict__ Av, long aBS, int lda,
    const void* __restrict__ Bv, long bBS, int ldb, float bscale,
    void* __restrict__ Cv, long cBS, int ldc,
    int M, int N, int Kd, int causal, int T_, int H_, int DH_)
{
  __shared__ unsigned short As[BMt * LDKp];
  __shared__ unsigned short Bs[BNt * LDKp];

  const int m0 = blockIdx.y * BMt;
  const int n0 = blockIdx.x * BNt;
  if (causal && n0 > m0 + (BMt - 1)) return;   // skip fully-masked score blocks
  const int bz = blockIdx.z;

  const int tid  = threadIdx.x;
  const int lane = tid & 31;
  const int wid  = tid >> 5;
  const int wm   = wid & 3;    // 4 row-bands of 32
  const int wn   = wid >> 2;   // 2 col-bands of 64

  const float*          Af = (const float*)Av;
  const unsigned short* Ah = (const unsigned short*)Av;
  const float*          Bf = (const float*)Bv;
  const unsigned short* Bh = (const unsigned short*)Bv;

  const size_t aoff = (size_t)bz * (size_t)aBS;
  const size_t boff = (size_t)bz * (size_t)bBS;
  const size_t coff = (size_t)bz * (size_t)cBS;

  v8f acc[2][4];
#pragma unroll
  for (int i = 0; i < 2; ++i)
#pragma unroll
    for (int j = 0; j < 4; ++j)
#pragma unroll
      for (int r = 0; r < 8; ++r) acc[i][j][r] = 0.0f;

  // global->LDS assignment: each thread moves 16 contiguous elems of one row
  const int trow = tid >> 1;
  const int tcol = (tid & 1) << 4;

  for (int k0 = 0; k0 < Kd; k0 += BKt) {
    { // ---- A tile (BMt x BKt) ----
      size_t g = aoff + (size_t)(m0 + trow) * (size_t)lda + (size_t)(k0 + tcol);
      unsigned short* dst = &As[trow * LDKp + tcol];
      if (AF32) {
        const float4* p = (const float4*)(Af + g);
#pragma unroll
        for (int i = 0; i < 4; ++i) {
          float4 f = p[i];
          dst[i*4+0] = f32_to_bf16(f.x);
          dst[i*4+1] = f32_to_bf16(f.y);
          dst[i*4+2] = f32_to_bf16(f.z);
          dst[i*4+3] = f32_to_bf16(f.w);
        }
        if (k0 + BKt < Kd) __builtin_prefetch((const void*)(Af + g + BKt), 0, 0);
      } else {
#if USE_ASYNC_LDS
        async_copy_b128(Ah + g,     dst);
        async_copy_b128(Ah + g + 8, dst + 8);
#else
        const uint4* p = (const uint4*)(Ah + g);
        ((uint4*)dst)[0] = p[0];
        ((uint4*)dst)[1] = p[1];
        if (k0 + BKt < Kd) __builtin_prefetch((const void*)(Ah + g + BKt), 0, 0);
#endif
      }
    }
    { // ---- B tile (BNt x BKt) ----
      size_t g = boff + (size_t)(n0 + trow) * (size_t)ldb + (size_t)(k0 + tcol);
      unsigned short* dst = &Bs[trow * LDKp + tcol];
      if (BF32) {
        const float4* p = (const float4*)(Bf + g);
#pragma unroll
        for (int i = 0; i < 4; ++i) {
          float4 f = p[i];
          dst[i*4+0] = f32_to_bf16(f.x * bscale);
          dst[i*4+1] = f32_to_bf16(f.y * bscale);
          dst[i*4+2] = f32_to_bf16(f.z * bscale);
          dst[i*4+3] = f32_to_bf16(f.w * bscale);
        }
        if (k0 + BKt < Kd) __builtin_prefetch((const void*)(Bf + g + BKt), 0, 0);
      } else {
#if USE_ASYNC_LDS
        async_copy_b128(Bh + g,     dst);
        async_copy_b128(Bh + g + 8, dst + 8);
#else
        const uint4* p = (const uint4*)(Bh + g);
        ((uint4*)dst)[0] = p[0];
        ((uint4*)dst)[1] = p[1];
        if (k0 + BKt < Kd) __builtin_prefetch((const void*)(Bh + g + BKt), 0, 0);
#endif
      }
    }
#if USE_ASYNC_LDS
    if constexpr (!AF32 || !BF32) {
      __builtin_amdgcn_s_wait_asynccnt(0);   // LDS writes landed (this wave)
    }
#endif
    __syncthreads();

    // ---- fragments per ISA layout; two ds_load_b128 each ----
    v16bf afr[2], bfr[4];
    {
      const int am = wm * 32 + (lane & 15);
      const int ak = (lane >> 4) << 3;          // lanes 16-31: +8
#pragma unroll
      for (int i = 0; i < 2; ++i) {
        const unsigned short* p = &As[(am + i * 16) * LDKp + ak];
        FragB16 u;
        u.u[0] = *(const uint4*)(p);            // k: ak .. ak+7
        u.u[1] = *(const uint4*)(p + 16);       // k: ak+16 .. ak+23
        afr[i] = u.v;
      }
      const int bn = wn * 64 + (lane & 15);
      const int bk = (lane >> 4) << 3;          // lanes 16-31: +8
#pragma unroll
      for (int j = 0; j < 4; ++j) {
        const unsigned short* p = &Bs[(bn + j * 16) * LDKp + bk];
        FragB16 u;
        u.u[0] = *(const uint4*)(p);            // k: bk .. bk+7
        u.u[1] = *(const uint4*)(p + 16);       // k: bk+16 .. bk+23
        bfr[j] = u.v;
      }
    }
#pragma unroll
    for (int i = 0; i < 2; ++i)
#pragma unroll
      for (int j = 0; j < 4; ++j)
        acc[i][j] = __builtin_amdgcn_wmma_f32_16x16x32_bf16(
            false, afr[i], false, bfr[j], (short)0, acc[i][j], false, false);

    __syncthreads();
  }

  // ---- epilogue ----
  float*          Cf = (float*)Cv;
  unsigned short* Ch = (unsigned short*)Cv;
  const int rbase = (lane >> 4) << 3;  // VGPR r -> M = r (+8 for lanes 16-31)
  const int cl    = lane & 15;
#pragma unroll
  for (int i = 0; i < 2; ++i) {
#pragma unroll
    for (int j = 0; j < 4; ++j) {
      const int tm = m0 + wm * 32 + i * 16 + rbase;
      const int tn = n0 + wn * 64 + j * 16 + cl;
#pragma unroll
      for (int r = 0; r < 8; ++r) {
        const int gm = tm + r;
        const int gn = tn;
        const float v = acc[i][j][r];
        if constexpr (MODE == 0) {
          Cf[coff + (size_t)gm * (size_t)ldc + gn] = v;
        } else if constexpr (MODE == 1) {
          const int b = gm / T_, t = gm % T_;
          const int h = gn / DH_, d = gn % DH_;
          Ch[((size_t)(b * H_ + h) * T_ + t) * DH_ + d] = f32_to_bf16(v);
        } else if constexpr (MODE == 2) {
          const int b = gm / T_, t = gm % T_;
          const int h = gn / DH_, d = gn % DH_;
          Ch[((size_t)(b * H_ + h) * DH_ + d) * T_ + t] = f32_to_bf16(v);
        } else { // MODE == 3
          const int b = bz / H_, h = bz % H_;
          Ch[(size_t)(b * T_ + gm) * (size_t)(H_ * DH_) + (size_t)h * DH_ + gn]
              = f32_to_bf16(v);
        }
      }
    }
  }
}

// ---------------------------------------------------------------------------
// Causal row softmax over S[bz][t][0..T), in place (fp32).
// Reads only j <= t; writes exp(s-max)/sum for j<=t and 0 for j>t.
// One 256-thread block per row.
// ---------------------------------------------------------------------------
__global__ __launch_bounds__(256) void softmax_causal_rows(float* __restrict__ S, int T)
{
  const int row = blockIdx.x;          // bz*T + t
  const int t   = row % T;
  float* s = S + (size_t)row * T;
  const int tid = threadIdx.x;
  __shared__ float red[256];

  float mx = -1e30f;
  for (int j = tid; j <= t; j += 256) mx = fmaxf(mx, s[j]);
  red[tid] = mx;
  __syncthreads();
  for (int st = 128; st > 0; st >>= 1) {
    if (tid < st) red[tid] = fmaxf(red[tid], red[tid + st]);
    __syncthreads();
  }
  mx = red[0];
  __syncthreads();

  float sum = 0.0f;
  for (int j = tid; j <= t; j += 256) sum += __expf(s[j] - mx);
  red[tid] = sum;
  __syncthreads();
  for (int st = 128; st > 0; st >>= 1) {
    if (tid < st) red[tid] += red[tid + st];
    __syncthreads();
  }
  const float inv = 1.0f / red[0];

  for (int j = tid; j <= t; j += 256) s[j] = __expf(s[j] - mx) * inv;
  for (int j = t + 1 + tid; j < T; j += 256) s[j] = 0.0f;
}

// ---------------------------------------------------------------------------
// Orchestration.  B=8, T=1024, K=512, H=8, DH=512, concat=4096, M=B*T=8192.
// Workspace (512 MB):
//   Qb  bf16 [64][1024][512]   64 MB
//   Kb  bf16 [64][1024][512]   64 MB
//   Vt  bf16 [64][512][1024]   64 MB   (transposed so PV GEMM is NT-form)
//   S   fp32 [64][1024][1024] 256 MB   (softmaxed in place)
//   AO  bf16 [8][1024][4096]   64 MB
// ---------------------------------------------------------------------------
extern "C" void kernel_launch(void* const* d_in, const int* in_sizes, int n_in,
                              void* d_out, int out_size, void* d_ws, size_t ws_size,
                              hipStream_t stream) {
  (void)in_sizes; (void)n_in; (void)out_size; (void)ws_size;
  const float* x  = (const float*)d_in[0];
  const float* Wq = (const float*)d_in[1];
  const float* Wk = (const float*)d_in[2];
  const float* Wv = (const float*)d_in[3];
  const float* Wo = (const float*)d_in[4];

  const long QKV = 64L * 1024L * 512L;     // 33,554,432 elems
  unsigned short* Qb = (unsigned short*)d_ws;
  unsigned short* Kb = Qb + QKV;
  unsigned short* Vt = Kb + QKV;
  float*          S  = (float*)(Vt + QKV);
  unsigned short* AO = (unsigned short*)(S + 64L * 1024L * 1024L);

  const float scale = 0.21022410381342863f;  // 512^-0.25 (applied to Wq and Wk)
  dim3 blk(256, 1, 1);

  // Q/K/V projections: M=8192, N=4096, K=512 (fp32 in, bf16 out, remapped)
  gemm_wmma<true, true, 1><<<dim3(32, 64, 1), blk, 0, stream>>>(
      x, 0L, 512, Wq, 0L, 512, scale, Qb, 0L, 0, 8192, 4096, 512, 0, 1024, 8, 512);
  gemm_wmma<true, true, 1><<<dim3(32, 64, 1), blk, 0, stream>>>(
      x, 0L, 512, Wk, 0L, 512, scale, Kb, 0L, 0, 8192, 4096, 512, 0, 1024, 8, 512);
  gemm_wmma<true, true, 2><<<dim3(32, 64, 1), blk, 0, stream>>>(
      x, 0L, 512, Wv, 0L, 512, 1.0f, Vt, 0L, 0, 8192, 4096, 512, 0, 1024, 8, 512);

  // Scores S = Q K^T (per bh batch; causal block-skip): M=N=1024, K=512
  gemm_wmma<false, false, 0><<<dim3(8, 8, 64), blk, 0, stream>>>(
      Qb, 524288L, 512, Kb, 524288L, 512, 1.0f,
      S, 1048576L, 1024, 1024, 1024, 512, 1, 1024, 8, 512);

  // Row softmax with causal mask (in place)
  softmax_causal_rows<<<dim3(65536, 1, 1), blk, 0, stream>>>(S, 1024);

  // O = P V : A = S fp32 (convert on load), B = Vt bf16; M=1024, N=512, K=1024
  gemm_wmma<true, false, 3><<<dim3(4, 8, 64), blk, 0, stream>>>(
      S, 1048576L, 1024, Vt, 524288L, 1024, 1.0f,
      AO, 0L, 0, 1024, 512, 1024, 0, 1024, 8, 512);

  // out = AO @ Wo^T : M=8192, N=512, K=4096, fp32 output
  gemm_wmma<false, true, 0><<<dim3(4, 64, 1), blk, 0, stream>>>(
      AO, 0L, 4096, Wo, 0L, 4096, 1.0f,
      d_out, 0L, 512, 8192, 512, 4096, 0, 1024, 8, 512);
}
